// SimpleFlowNet_51324859187624
// MI455X (gfx1250) — matrix operand order repaired and checked by reading, not compiled, not run on hardware
//
#include <hip/hip_runtime.h>
#include <cstddef>
#include <cstdint>

// ---------------------------------------------------------------------------
// CDNA5 (gfx1250, wave32) WMMA types: v_wmma_f32_16x16x32_bf16
// ---------------------------------------------------------------------------
typedef __bf16 bf16_t;
typedef __bf16 v16bf __attribute__((ext_vector_type(16)));
typedef __bf16 v8bf  __attribute__((ext_vector_type(8)));
typedef float  v8f   __attribute__((ext_vector_type(8)));
typedef int    v4i_  __attribute__((vector_size(16)));   // matches builtin proto

#define BN_EPS   1e-5f
#define NORM_EPS 1e-12f

#if defined(__has_builtin)
#if __has_builtin(__builtin_amdgcn_global_load_async_to_lds_b128)
#define USE_ASYNC_LDS 1
#endif
#endif

// ---------------------------------------------------------------------------
// Weight repack: w[Cout,Cin,KH,KW] fp32 -> wp[kt][CoutP][32] bf16
// kt = (ky,kx,cit), cit innermost. Zero-pad ci>=Cin / co>=Cout. Matches the
// A-fragment 16x32 bf16 layout AND the NHWC channel tiling of activations.
// ---------------------------------------------------------------------------
__global__ void pack_w_kernel(const float* __restrict__ w, bf16_t* __restrict__ wp,
                              int Cout, int CoutP, int Cin, int KH, int KW) {
  const int citiles = (Cin + 31) >> 5;
  const long long total = (long long)KH * KW * citiles * CoutP * 32;
  long long t = (long long)blockIdx.x * blockDim.x + threadIdx.x;
  if (t >= total) return;
  int kk = (int)(t & 31);
  long long r = t >> 5;
  int co = (int)(r % CoutP);
  int kt = (int)(r / CoutP);
  int cit = kt % citiles;
  int sp  = kt / citiles;
  int ky = sp / KW, kx = sp - ky * KW;
  int ci = cit * 32 + kk;
  float v = 0.f;
  if (co < Cout && ci < Cin)
    v = w[(((long long)co * Cin + ci) * KH + ky) * KW + kx];
  wp[t] = (bf16_t)v;
}

// Fold conv-bias + BN into per-channel scale/shift over CoutP (zeros in pad,
// so padded output channels compute to exactly 0 with no extra guards).
__global__ void fold_bn_kernel(const float* __restrict__ cb, const float* __restrict__ g,
                               const float* __restrict__ bb, const float* __restrict__ m,
                               const float* __restrict__ v, float* __restrict__ scale,
                               float* __restrict__ shift, int Cout, int CoutP, int hasbn) {
  int c = blockIdx.x * blockDim.x + threadIdx.x;
  if (c >= CoutP) return;
  if (c >= Cout) { scale[c] = 0.f; shift[c] = 0.f; return; }
  if (hasbn) {
    float inv = g[c] * rsqrtf(v[c] + BN_EPS);
    scale[c] = inv;
    shift[c] = bb[c] - m[c] * inv + cb[c] * inv;
  } else {
    scale[c] = 1.f;
    shift[c] = cb[c];
  }
}

// Image NCHW fp32 [B,3,H,W] -> NHWC bf16 [B,H,W,32], zero channel pad.
__global__ void img_to_nhwc_kernel(const float* __restrict__ img, bf16_t* __restrict__ out,
                                   int B, int H, int W) {
  long long idx = (long long)blockIdx.x * blockDim.x + threadIdx.x;
  long long total = (long long)B * H * W * 32;
  if (idx >= total) return;
  int c = (int)(idx & 31);
  long long p = idx >> 5;
  int x = (int)(p % W); long long r = p / W;
  int y = (int)(r % H); int b = (int)(r / H);
  float v = (c < 3) ? img[(((long long)b * 3 + c) * H + y) * W + x] : 0.f;
  out[idx] = (bf16_t)v;
}

// ---------------------------------------------------------------------------
// Implicit-GEMM conv on WMMA, NHWC bf16 activations, fp32 accumulate.
//   Block: 8 waves share one 16-channel co-tile; weights (KT x 16 x 32 bf16)
//   staged once into LDS via async global->LDS copies, then each wave computes
//   a 16(co) x 64(pixel) tile: one A fragment (2x ds_read_b128) feeds FOUR
//   back-to-back wmmas. B fragments are contiguous 2x16B NHWC loads per pixel.
// Epilogue: scale/shift (+bf16 NHWC residual)(+relu) -> packed bf16 b128 store,
// or NCHW fp32 store mode (+fp32 residual) for the final flow conv.
// ---------------------------------------------------------------------------
__global__ __launch_bounds__(256)
void conv_wmma_kernel(const bf16_t* __restrict__ in, const bf16_t* __restrict__ wp,
                      const float* __restrict__ scale, const float* __restrict__ shift,
                      const bf16_t* __restrict__ resb, const float* __restrict__ resf,
                      bf16_t* __restrict__ outb, float* __restrict__ outf,
                      int B, int CinP, int Hin, int Win,
                      int Cout, int CoutP, int Hout, int Wout,
                      int KH, int KW, int stride, int pad, int relu, int nchw_out) {
  extern __shared__ bf16_t sW[];                   // [KT][16][32] bf16
  const int lane = threadIdx.x & 31;
  const int wave = threadIdx.x >> 5;
  const int HWo  = Hout * Wout;
  const int NPix = B * HWo;
  const int coTiles = CoutP >> 4;
  const int ct = blockIdx.x % coTiles;             // co-tile shared by block
  const int pg = blockIdx.x / coTiles;             // 512-pixel group
  const int citiles = CinP >> 5;
  const int KT = KH * KW * citiles;

  // ---- stage this co-tile's weights into LDS (async copy, 16B chunks) ----
  for (int c = threadIdx.x; c < KT * 64; c += 256) {
    const int kt   = c >> 6;
    const int rem  = c & 63;
    const int row  = rem >> 2;
    const int part = rem & 3;
    const bf16_t* gp = wp + ((long long)kt * CoutP + ct * 16 + row) * 32 + part * 8;
    bf16_t* lp = sW + (kt * 16 + row) * 32 + part * 8;
#ifdef USE_ASYNC_LDS
    __builtin_amdgcn_global_load_async_to_lds_b128(
        (__attribute__((address_space(1))) v4i_*)gp,
        (__attribute__((address_space(3))) v4i_*)lp, 0, 0);
#else
    *(v8bf*)lp = *(const v8bf*)gp;
#endif
  }
#ifdef USE_ASYNC_LDS
#if __has_builtin(__builtin_amdgcn_s_wait_asynccnt)
  __builtin_amdgcn_s_wait_asynccnt(0);
#else
  asm volatile("s_wait_asynccnt 0x0" ::: "memory");
#endif
#endif
  __syncthreads();

  const int lhalf = lane >> 4, lmod = lane & 15;
  const int pbase = (pg * 8 + wave) * 64;          // wave-uniform
  if (pbase >= NPix) return;                       // after the barrier: safe

  // decode 4 pixels per lane
  int qn[4], qoy[4], qox[4]; bool qv[4]; int qq[4];
#pragma unroll
  for (int i = 0; i < 4; ++i) {
    const int p = pbase + i * 16 + lmod;
    qv[i] = p < NPix;
    const int q = qv[i] ? p : 0;
    qq[i] = q;
    qn[i] = q / HWo;
    const int rr = q - qn[i] * HWo;
    qoy[i] = rr / Wout;
    qox[i] = rr - qoy[i] * Wout;
  }

  const bf16_t* wl = sW + lmod * 32 + lhalf * 16;  // A row base in LDS
  v8f acc[4] = {v8f{}, v8f{}, v8f{}, v8f{}};
  int cit = 0, kx = 0, ky = 0;
  for (int kt = 0; kt < KT; ++kt) {
    // A fragment from LDS: two ds_read_b128
    v8bf alo = *(const v8bf*)(wl + kt * 512);
    v8bf ahi = *(const v8bf*)(wl + kt * 512 + 8);
    v16bf afrag;
#pragma unroll
    for (int j = 0; j < 8; ++j) { afrag[j] = alo[j]; afrag[j + 8] = ahi[j]; }

    const int kbase = cit * 32 + lhalf * 16;       // K = kbase + j (NHWC contig)
#pragma unroll
    for (int i = 0; i < 4; ++i) {
      const int iy = qoy[i] * stride - pad + ky;
      const int ix = qox[i] * stride - pad + kx;
      v8bf blo = {}, bhi = {};
      if (qv[i] && (unsigned)iy < (unsigned)Hin && (unsigned)ix < (unsigned)Win) {
        const bf16_t* bp = in + ((long long)(qn[i] * Hin + iy) * Win + ix) * CinP + kbase;
        blo = *(const v8bf*)bp; bhi = *(const v8bf*)(bp + 8);
      }
      v16bf bfr;
#pragma unroll
      for (int j = 0; j < 8; ++j) { bfr[j] = blo[j]; bfr[j + 8] = bhi[j]; }
      acc[i] = __builtin_amdgcn_wmma_f32_16x16x32_bf16(false, afrag, false, bfr,
                                                       (short)0, acc[i], false, false);
    }
    if (++cit == citiles) { cit = 0; if (++kx == KW) { kx = 0; ++ky; } }
  }

  const int cbase = ct * 16 + lhalf * 8;           // 8 contiguous output channels
  const v8f sv = *(const v8f*)(scale + cbase);
  const v8f hv = *(const v8f*)(shift + cbase);

  if (!nchw_out) {
#pragma unroll
    for (int i = 0; i < 4; ++i) {
      if (!qv[i]) continue;
      v8f val;
#pragma unroll
      for (int r = 0; r < 8; ++r) val[r] = sv[r] * acc[i][r] + hv[r];
      if (resb) {
        v8bf rv = *(const v8bf*)(resb + (long long)qq[i] * CoutP + cbase);
#pragma unroll
        for (int r = 0; r < 8; ++r) val[r] += (float)rv[r];
      }
      v8bf ov;
#pragma unroll
      for (int r = 0; r < 8; ++r) ov[r] = (bf16_t)(relu ? fmaxf(val[r], 0.f) : val[r]);
      *(v8bf*)(outb + (long long)qq[i] * CoutP + cbase) = ov;
    }
  } else {
#pragma unroll
    for (int i = 0; i < 4; ++i) {
#pragma unroll
      for (int r = 0; r < 8; ++r) {
        const int co = cbase + r;
        if (qv[i] && co < Cout) {
          float v = sv[r] * acc[i][r] + hv[r];
          const long long oi =
              (((long long)qn[i] * Cout + co) * Hout + qoy[i]) * Wout + qox[i];
          if (resf) v += resf[oi];
          outf[oi] = relu ? fmaxf(v, 0.f) : v;
        }
      }
    }
  }
}

// ---------------------------------------------------------------------------
// L2-normalize over channels (NHWC, unit-stride reduction per pixel).
// ---------------------------------------------------------------------------
__global__ void norm_kernel(const bf16_t* __restrict__ f, bf16_t* __restrict__ fn,
                            long long NPix, int C) {
  long long idx = (long long)blockIdx.x * blockDim.x + threadIdx.x;
  if (idx >= NPix) return;
  const bf16_t* src = f + idx * C;
  float s = 0.f;
  for (int c = 0; c < C; ++c) { float v = (float)src[c]; s += v * v; }
  float inv = 1.f / fmaxf(sqrtf(s), NORM_EPS);
  bf16_t* dst = fn + idx * C;
  for (int c = 0; c < C; ++c) dst[c] = (bf16_t)((float)src[c] * inv);
}

// Correlation volume, NHWC in, writes channels 0..80 of 96-padded decoder input.
__global__ void corr_kernel(const bf16_t* __restrict__ f1, const bf16_t* __restrict__ f2,
                            bf16_t* __restrict__ xcat, int B, int C, int H, int W, int CtotP) {
  long long idx = (long long)blockIdx.x * blockDim.x + threadIdx.x;
  long long total = (long long)B * 81 * H * W;
  if (idx >= total) return;
  int x = (int)(idx % W); long long r = idx / W;
  int y = (int)(r % H);  r /= H;
  int d = (int)(r % 81); int b = (int)(r / 81);
  int dy = d / 9 - 4, dx = d % 9 - 4;
  int y2 = y + dy, x2 = x + dx;
  float s = 0.f;
  if ((unsigned)y2 < (unsigned)H && (unsigned)x2 < (unsigned)W) {
    const bf16_t* a  = f1 + ((long long)(b * H + y)  * W + x)  * C;
    const bf16_t* bb = f2 + ((long long)(b * H + y2) * W + x2) * C;
    for (int c = 0; c < C; ++c) s += (float)a[c] * (float)bb[c];
  }
  xcat[((long long)(b * H + y) * W + x) * CtotP + d] = (bf16_t)s;
}

// Bilinear (align_corners) 2x upsample of flow (NCHW fp32 in), value*2.
// Writes fp32 NCHW flow_up (for the fc residual) and channels 81..82 of concat.
__global__ void upsample_kernel(const float* __restrict__ flo, float* __restrict__ fup,
                                bf16_t* __restrict__ xcat, int B, int H, int W, int CtotP) {
  const int H2 = 2 * H, W2 = 2 * W;
  long long idx = (long long)blockIdx.x * blockDim.x + threadIdx.x;
  long long total = (long long)B * 2 * H2 * W2;
  if (idx >= total) return;
  int x = (int)(idx % W2); long long r = idx / W2;
  int y = (int)(r % H2);  r /= H2;
  int c = (int)(r % 2);   int b = (int)(r / 2);
  float sy = (H > 1) ? (float)(H - 1) / (float)(H2 - 1) : 0.f;
  float sx = (W > 1) ? (float)(W - 1) / (float)(W2 - 1) : 0.f;
  float fy = y * sy, fx = x * sx;
  int y0 = (int)floorf(fy); int y1 = (y0 + 1 < H) ? y0 + 1 : H - 1;
  int x0 = (int)floorf(fx); int x1 = (x0 + 1 < W) ? x0 + 1 : W - 1;
  float wy = fy - y0, wx = fx - x0;
  const float* src = flo + ((long long)b * 2 + c) * H * W;
  float v = (1.f - wy) * ((1.f - wx) * src[y0 * W + x0] + wx * src[y0 * W + x1]) +
            wy        * ((1.f - wx) * src[y1 * W + x0] + wx * src[y1 * W + x1]);
  v *= 2.f;
  fup[(((long long)b * 2 + c) * H2 + y) * W2 + x] = v;
  xcat[((long long)(b * H2 + y) * W2 + x) * CtotP + 81 + c] = (bf16_t)v;
}

// Zero channels [c0, CP) of an NHWC tensor (keeps wmma inputs NaN-free).
__global__ void zero_pad_kernel(bf16_t* __restrict__ t, long long NPix, int CP, int c0) {
  const int nc = CP - c0;
  long long idx = (long long)blockIdx.x * blockDim.x + threadIdx.x;
  if (idx >= NPix * nc) return;
  long long p = idx / nc;
  int c = (int)(idx - p * nc);
  t[p * CP + c0 + c] = (bf16_t)0.f;
}

// ---------------------------------------------------------------------------
// Host orchestration. d_in leaf order (depth-first dict insertion order):
//  0 img1, 1 img2, fe.conv1{2,3}, fe.bn1{4..7},
//  rb1{8..19}, rb2{20..37}, rb3{38..49}, rb4{50..67}, rb5{68..79},
//  dec{80..87}
// ---------------------------------------------------------------------------
struct ConvSpec { int wi, bi, gi, Cin, Cout, KH; };
static const ConvSpec g_specs[17] = {
  {2, 3, 4,   3,  32, 7},   // 0  conv1+bn1
  {8, 9, 10, 32,  32, 3},   // 1  rb1.c1
  {14,15,16, 32,  32, 3},   // 2  rb1.c2
  {20,21,22, 32,  64, 3},   // 3  rb2.c1
  {26,27,28, 64,  64, 3},   // 4  rb2.c2
  {32,33,34, 32,  64, 1},   // 5  rb2.sc
  {38,39,40, 64,  64, 3},   // 6  rb3.c1
  {44,45,46, 64,  64, 3},   // 7  rb3.c2
  {50,51,52, 64, 128, 3},   // 8  rb4.c1
  {56,57,58,128, 128, 3},   // 9  rb4.c2
  {62,63,64, 64, 128, 1},   // 10 rb4.sc
  {68,69,70,128, 128, 3},   // 11 rb5.c1
  {74,75,76,128, 128, 3},   // 12 rb5.c2
  {80,81,-1, 83, 128, 3},   // 13 dec.c1
  {82,83,-1,128,  64, 3},   // 14 dec.c2
  {84,85,-1, 64,  32, 3},   // 15 dec.c3
  {86,87,-1, 32,   2, 3},   // 16 dec.fc
};

extern "C" void kernel_launch(void* const* d_in, const int* in_sizes, int n_in,
                              void* d_out, int out_size, void* d_ws, size_t ws_size,
                              hipStream_t stream) {
  (void)in_sizes; (void)n_in; (void)out_size;
  const int B = 4;
  const int N0 = 192 * 192, N1 = 96 * 96, N2 = 48 * 48;

  uint8_t* base = (uint8_t*)d_ws;
  size_t cur = 0;
  auto alloc = [&](size_t bytes) -> void* {
    void* p = base + cur;
    cur += (bytes + 255) & ~(size_t)255;
    return p;
  };

  // --- workspace layout (deterministic every call) ---
  bf16_t* wp[17]; float* scp[17]; float* shp[17]; long long pelems[17];
  for (int li = 0; li < 17; ++li) {
    const ConvSpec& S = g_specs[li];
    const int CoutP = (S.Cout + 31) & ~31;
    const int citiles = (S.Cin + 31) >> 5;
    pelems[li] = (long long)S.KH * S.KH * citiles * CoutP * 32;
    wp[li]  = (bf16_t*)alloc((size_t)pelems[li] * sizeof(bf16_t));
    scp[li] = (float*)alloc((size_t)CoutP * sizeof(float));
    shp[li] = (float*)alloc((size_t)CoutP * sizeof(float));
  }
  // NHWC bf16 activations (channels padded to mult of 32; features are exact)
  bf16_t* imgb = (bf16_t*)alloc((size_t)B * 384 * 384 * 32 * 2);
  bf16_t* f0i[2]; bf16_t* f1i[2]; bf16_t* f2i[2];
  for (int im = 0; im < 2; ++im) {
    f0i[im] = (bf16_t*)alloc((size_t)B * N0 * 32 * 2);
    f1i[im] = (bf16_t*)alloc((size_t)B * N1 * 64 * 2);
    f2i[im] = (bf16_t*)alloc((size_t)B * N2 * 128 * 2);
  }
  bf16_t* bufA = (bf16_t*)alloc((size_t)B * N0 * 32 * 2);
  bf16_t* bufB = (bf16_t*)alloc((size_t)B * N0 * 32 * 2);
  bf16_t* bufC = (bf16_t*)alloc((size_t)B * N1 * 64 * 2);
  bf16_t* bufD = (bf16_t*)alloc((size_t)B * N1 * 64 * 2);
  bf16_t* bufE = (bf16_t*)alloc((size_t)B * N1 * 64 * 2);
  bf16_t* bufF = (bf16_t*)alloc((size_t)B * N2 * 128 * 2);
  bf16_t* bufG = (bf16_t*)alloc((size_t)B * N2 * 128 * 2);
  bf16_t* bufH = (bf16_t*)alloc((size_t)B * N2 * 128 * 2);
  bf16_t* nf1  = (bf16_t*)alloc((size_t)B * N0 * 32 * 2);  // max over levels
  bf16_t* nf2  = (bf16_t*)alloc((size_t)B * N0 * 32 * 2);
  bf16_t* xin  = (bf16_t*)alloc((size_t)B * N0 * 96 * 2);  // 83 -> pad 96
  bf16_t* t1   = (bf16_t*)alloc((size_t)B * N0 * 128 * 2);
  bf16_t* t2   = (bf16_t*)alloc((size_t)B * N0 * 64 * 2);
  bf16_t* t3   = (bf16_t*)alloc((size_t)B * N0 * 32 * 2);
  float*  fup  = (float*)alloc((size_t)B * 2 * N0 * 4);    // NCHW fp32
  if (cur > ws_size) return;

  // --- phase 0: pack weights + fold BN (tiny) ---
  for (int li = 0; li < 17; ++li) {
    const ConvSpec& S = g_specs[li];
    const int CoutP = (S.Cout + 31) & ~31;
    pack_w_kernel<<<(int)((pelems[li] + 255) / 256), 256, 0, stream>>>(
        (const float*)d_in[S.wi], wp[li], S.Cout, CoutP, S.Cin, S.KH, S.KH);
    const float* g = (S.gi >= 0) ? (const float*)d_in[S.gi]     : nullptr;
    const float* b = (S.gi >= 0) ? (const float*)d_in[S.gi + 1] : nullptr;
    const float* m = (S.gi >= 0) ? (const float*)d_in[S.gi + 2] : nullptr;
    const float* v = (S.gi >= 0) ? (const float*)d_in[S.gi + 3] : nullptr;
    fold_bn_kernel<<<(CoutP + 255) / 256, 256, 0, stream>>>(
        (const float*)d_in[S.bi], g, b, m, v, scp[li], shp[li], S.Cout, CoutP, S.gi >= 0);
  }

  auto conv = [&](int li, const bf16_t* in, bf16_t* outb, float* outf,
                  const bf16_t* resb, const float* resf,
                  int Hin, int stride, int pad, int relu) {
    const ConvSpec& S = g_specs[li];
    const int CinP  = (S.Cin + 31) & ~31;
    const int CoutP = (S.Cout + 31) & ~31;
    const int Hout = (Hin + 2 * pad - S.KH) / stride + 1;
    const int NPix = B * Hout * Hout;
    const int KT = S.KH * S.KH * (CinP >> 5);
    const int coTiles = CoutP >> 4;
    const int pixBlocks = (NPix + 511) / 512;
    const int blocks = coTiles * pixBlocks;
    const size_t shmem = (size_t)KT * 16 * 32 * sizeof(bf16_t);
    conv_wmma_kernel<<<blocks, 256, shmem, stream>>>(
        in, wp[li], scp[li], shp[li], resb, resf, outb, outf,
        B, CinP, Hin, Hin, S.Cout, CoutP, Hout, Hout,
        S.KH, S.KH, stride, pad, relu, outf != nullptr);
  };

  // --- phase 1: twin feature extraction (NHWC bf16 throughout) ---
  for (int im = 0; im < 2; ++im) {
    long long it = (long long)B * 384 * 384 * 32;
    img_to_nhwc_kernel<<<(int)((it + 255) / 256), 256, 0, stream>>>(
        (const float*)d_in[im], imgb, B, 384, 384);
    conv(0, imgb, bufA, nullptr, nullptr, nullptr, 384, 2, 3, 1); // conv1+bn1
    conv(1, bufA, bufB, nullptr, nullptr, nullptr, 192, 1, 1, 1); // rb1.c1
    conv(2, bufB, f0i[im], nullptr, bufA, nullptr, 192, 1, 1, 1); // rb1.c2+res
    conv(3, f0i[im], bufC, nullptr, nullptr, nullptr, 192, 2, 1, 1); // rb2.c1
    conv(5, f0i[im], bufD, nullptr, nullptr, nullptr, 192, 2, 0, 0); // rb2.sc
    conv(4, bufC, bufE, nullptr, bufD, nullptr, 96, 1, 1, 1);     // rb2.c2+sc
    conv(6, bufE, bufC, nullptr, nullptr, nullptr, 96, 1, 1, 1);  // rb3.c1
    conv(7, bufC, f1i[im], nullptr, bufE, nullptr, 96, 1, 1, 1);  // rb3.c2+res
    conv(8, f1i[im], bufF, nullptr, nullptr, nullptr, 96, 2, 1, 1);  // rb4.c1
    conv(10, f1i[im], bufG, nullptr, nullptr, nullptr, 96, 2, 0, 0); // rb4.sc
    conv(9, bufF, bufH, nullptr, bufG, nullptr, 48, 1, 1, 1);     // rb4.c2+sc
    conv(11, bufH, bufF, nullptr, nullptr, nullptr, 48, 1, 1, 1); // rb5.c1
    conv(12, bufF, f2i[im], nullptr, bufH, nullptr, 48, 1, 1, 1); // rb5.c2+res
  }

  // --- phase 2: coarse-to-fine decode (levels i=2,1,0) ---
  const int Hs[3] = {192, 96, 48};
  const int Cs[3] = {32, 64, 128};
  bf16_t* feat1[3] = {f0i[0], f1i[0], f2i[0]};
  bf16_t* feat2[3] = {f0i[1], f1i[1], f2i[1]};
  const long long outOff[3] = {(long long)B*2*N2 + (long long)B*2*N1,
                               (long long)B*2*N2,
                               0};
  const float* prevFlow = nullptr;
  for (int i = 2; i >= 0; --i) {
    const int H = Hs[i], C = Cs[i];
    const long long NPix = (long long)B * H * H;
    norm_kernel<<<(int)((NPix + 255) / 256), 256, 0, stream>>>(feat1[i], nf1, NPix, C);
    norm_kernel<<<(int)((NPix + 255) / 256), 256, 0, stream>>>(feat2[i], nf2, NPix, C);
    long long ct = NPix * 81;
    corr_kernel<<<(int)((ct + 255) / 256), 256, 0, stream>>>(nf1, nf2, xin, B, C, H, H, 96);
    if (i == 2) {
      long long zt = NPix * (96 - 81);
      zero_pad_kernel<<<(int)((zt + 255) / 256), 256, 0, stream>>>(xin, NPix, 96, 81);
    } else {
      long long zt = NPix * (96 - 83);
      zero_pad_kernel<<<(int)((zt + 255) / 256), 256, 0, stream>>>(xin, NPix, 96, 83);
      const int Hc = Hs[i + 1];
      long long ut = NPix * 2;
      upsample_kernel<<<(int)((ut + 255) / 256), 256, 0, stream>>>(prevFlow, fup, xin, B, Hc, Hc, 96);
    }
    conv(13, xin, t1, nullptr, nullptr, nullptr, H, 1, 1, 1);
    conv(14, t1,  t2, nullptr, nullptr, nullptr, H, 1, 1, 1);
    conv(15, t2,  t3, nullptr, nullptr, nullptr, H, 1, 1, 1);
    float* flow = (float*)d_out + outOff[i];
    conv(16, t3, nullptr, flow, nullptr, (i == 2) ? nullptr : fup, H, 1, 1, 0);
    prevFlow = flow;
  }
}
